// TriangularSelfAttentionBlock_67757404061860
// MI455X (gfx1250) — matrix operand order repaired and checked by reading, not compiled, not run on hardware
//
#include <hip/hip_runtime.h>
#include <hip/hip_bf16.h>
#include <cstdint>
#include <cstddef>

// ---------------------------------------------------------------------------
// TriangularSelfAttentionBlock for MI455X (gfx1250), wave32 + WMMA.
// All heavy math runs through v_wmma_f32_16x16x32_f16 (f16 in, f32 acc).
// ---------------------------------------------------------------------------

typedef __attribute__((ext_vector_type(16))) _Float16 v16h;
typedef __attribute__((ext_vector_type(8)))  _Float16 v8h;
typedef __attribute__((ext_vector_type(8)))  float    v8f;

#define L_SEQ 256
#define LLPAIR 65536   // L*L
#define CZ_DIM 128
#define CS_DIM 1024
#define QK_SCALE 0.17677669529663687f   // 32^-0.5

// ------------------------------- device helpers ----------------------------

__device__ __forceinline__ float sigm(float x) { return 1.0f / (1.0f + expf(-x)); }

// A-matrix fragment, 16x32 f16 (ISA 7.12.2): lane m = lane&15,
// kbase = 8*(lane>=16); elements 0..7 -> K=kbase..kbase+7, 8..15 -> kbase+16..23.
__device__ __forceinline__ v16h load_fragA(const _Float16* __restrict__ base, int ld,
                                           int row, int kk, int lane) {
  int kb = kk + ((lane >> 4) << 3);
  const _Float16* p = base + (size_t)row * ld + kb;
  v8h lo = *(const v8h*)(p);
  v8h hi = *(const v8h*)(p + 16);
  v16h r;
#pragma unroll
  for (int i = 0; i < 8; ++i) { r[i] = lo[i]; r[i + 8] = hi[i]; }
  return r;
}

// B-matrix fragment, 32x16 f16 stored pre-transposed as Bt[N][K]:
// lane n = lane&15, kbase = 8*(lane>=16); elements 0..7 -> K=kbase..kbase+7, 8..15 -> kbase+16..23.
__device__ __forceinline__ v16h load_fragB(const _Float16* __restrict__ bt, int ld,
                                           int n, int kk, int lane) {
  int kb = kk + ((lane >> 4) << 3);
  const _Float16* p = bt + (size_t)n * ld + kb;
  v8h lo = *(const v8h*)(p);
  v8h hi = *(const v8h*)(p + 16);
  v16h r;
#pragma unroll
  for (int i = 0; i < 8; ++i) { r[i] = lo[i]; r[i + 8] = hi[i]; }
  return r;
}

__device__ __forceinline__ v8f wmma16(v16h a, v16h b, v8f c) {
  return __builtin_amdgcn_wmma_f32_16x16x32_f16(false, a, false, b, (short)0, c,
                                                false, false);
}

// ------------------------------- GEMM kernel -------------------------------
// C[b][M][N] = act(A[b][M][K] * Bt[b][N][K]^T + bias) (+ addsrc)
// act: 0 = none, 1 = relu, 2 = sigmoid.  If C16 != null, store f16 instead.
// Wave id is scalarized (readfirstlane) so tile guards are SALU branches and
// the hot loop is straight-line loads + back-to-back WMMA with EXEC all-ones.
__global__ void gemm_wmma_kernel(const _Float16* __restrict__ A, size_t sA,
                                 const _Float16* __restrict__ Bt, size_t sB,
                                 const float* __restrict__ bias,
                                 const float* __restrict__ addsrc,
                                 float* __restrict__ C,
                                 _Float16* __restrict__ C16, size_t sC,
                                 int M, int N, int K, int act) {
  const int lane = threadIdx.x & 31;
  const int wave = __builtin_amdgcn_readfirstlane((int)(threadIdx.x >> 5));  // 0..3, SGPR
  const int wm = wave >> 1, wn = wave & 1;  // 2x2 waves -> 64x64 block tile
  const int m0 = blockIdx.y * 64 + wm * 32;
  const int n0 = blockIdx.x * 64 + wn * 32;
  const _Float16* Ab = A + (size_t)blockIdx.z * sA;
  const _Float16* Bb = Bt + (size_t)blockIdx.z * sB;

  const bool nok0 = (n0 < N);        // uniform (SGPR-derived)
  const bool nok1 = (n0 + 16 < N);   // uniform

  v8f acc00 = {}, acc01 = {}, acc10 = {}, acc11 = {};

  const int ar0 = m0 + (lane & 15);
  const int ar1 = m0 + 16 + (lane & 15);
  const int bn0 = n0 + (lane & 15);
  const int bn1 = n0 + 16 + (lane & 15);

  if (nok1) {
    // fast path: both n-tiles valid -> 8 loads + 4 WMMAs per K step, no branches
    for (int kk = 0; kk < K; kk += 32) {
      v16h a0 = load_fragA(Ab, K, ar0, kk, lane);
      v16h a1 = load_fragA(Ab, K, ar1, kk, lane);
      v16h b0 = load_fragB(Bb, K, bn0, kk, lane);
      v16h b1 = load_fragB(Bb, K, bn1, kk, lane);
      acc00 = wmma16(a0, b0, acc00);
      acc10 = wmma16(a1, b0, acc10);
      acc01 = wmma16(a0, b1, acc01);
      acc11 = wmma16(a1, b1, acc11);
    }
  } else if (nok0) {
    for (int kk = 0; kk < K; kk += 32) {
      v16h a0 = load_fragA(Ab, K, ar0, kk, lane);
      v16h a1 = load_fragA(Ab, K, ar1, kk, lane);
      v16h b0 = load_fragB(Bb, K, bn0, kk, lane);
      acc00 = wmma16(a0, b0, acc00);
      acc10 = wmma16(a1, b0, acc10);
    }
  } else {
    return;  // whole wave's tiles out of range (uniform)
  }

  float* Cb = C ? (C + (size_t)blockIdx.z * sC) : nullptr;
  _Float16* Hb = C16 ? (C16 + (size_t)blockIdx.z * sC) : nullptr;

  v8f accs[2][2] = {{acc00, acc01}, {acc10, acc11}};
#pragma unroll
  for (int mt = 0; mt < 2; ++mt) {
#pragma unroll
    for (int nt = 0; nt < 2; ++nt) {
      if (nt == 1 && !nok1) continue;
      const int mmb = m0 + mt * 16 + ((lane >> 4) << 3);  // C layout: row = r + 8*(lane>=16)
      const int nn = n0 + nt * 16 + (lane & 15);
      const float bv = bias ? bias[nn] : 0.0f;
#pragma unroll
      for (int r = 0; r < 8; ++r) {
        float v = accs[mt][nt][r] + bv;
        if (act == 1) v = fmaxf(v, 0.0f);
        else if (act == 2) v = sigm(v);
        const size_t idx = (size_t)(mmb + r) * N + nn;
        if (addsrc) v += addsrc[idx];
        if (Hb) Hb[idx] = (_Float16)v;
        else    Cb[idx] = v;
      }
    }
  }
}

// --------------------------- weight convert (W[K,N] -> Wt f16 [Npad,K]) -----
__global__ void wconv_kernel(const float* __restrict__ W, _Float16* __restrict__ Wt,
                             int K, int N, int Npad, size_t total) {
  size_t i = (size_t)blockIdx.x * blockDim.x + threadIdx.x;
  if (i >= total) return;
  int n = (int)(i / K), k = (int)(i % K);
  Wt[i] = (n < N) ? (_Float16)W[(size_t)k * N + n] : (_Float16)0.0f;
}

// --------------------------- LayerNorm (rows x D) -> f16 --------------------
// swap: read pair-row (j,i) instead of (i,j).  chanMajor: X is [c][rows].
__global__ void ln_kernel(const float* __restrict__ X, const float* __restrict__ g,
                          const float* __restrict__ b, _Float16* __restrict__ Y,
                          int D, int rows, int Lp, int swap, int chanMajor) {
  const int r = blockIdx.x;
  const int sr = swap ? ((r % Lp) * Lp + (r / Lp)) : r;
  const int t = threadIdx.x;
  __shared__ float red[128];

  float s = 0.0f;
  for (int c = t; c < D; c += 128) {
    size_t idx = chanMajor ? ((size_t)c * rows + sr) : ((size_t)sr * D + c);
    s += X[idx];
  }
  red[t] = s; __syncthreads();
  for (int o = 64; o > 0; o >>= 1) { if (t < o) red[t] += red[t + o]; __syncthreads(); }
  const float mu = red[0] / D; __syncthreads();

  float vs = 0.0f;
  for (int c = t; c < D; c += 128) {
    size_t idx = chanMajor ? ((size_t)c * rows + sr) : ((size_t)sr * D + c);
    float d = X[idx] - mu; vs += d * d;
  }
  red[t] = vs; __syncthreads();
  for (int o = 64; o > 0; o >>= 1) { if (t < o) red[t] += red[t + o]; __syncthreads(); }
  const float rinv = rsqrtf(red[0] / D + 1e-5f);

  for (int c = t; c < D; c += 128) {
    size_t idx = chanMajor ? ((size_t)c * rows + sr) : ((size_t)sr * D + c);
    Y[(size_t)r * D + c] = (_Float16)((X[idx] - mu) * rinv * g[c] + b[c]);
  }
}

// ------------------------- softmax: sequence attention ----------------------
// sc [H][L][L]; bias [L*L, H] (pair->seq bias); mask [L]; probs f16 [H][L][L]
__global__ void softmax_seq_kernel(const float* __restrict__ sc,
                                   const float* __restrict__ bias,
                                   const float* __restrict__ mask,
                                   _Float16* __restrict__ probs) {
  const int q = blockIdx.x, h = blockIdx.y, k = threadIdx.x;
  const size_t rowo = ((size_t)h * L_SEQ + q) * L_SEQ;
  float v = sc[rowo + k] + bias[((size_t)q * L_SEQ + k) * 32 + h];
  if (!(mask[k] > 0.0f)) v = -1e9f;
  __shared__ float red[256];
  red[k] = v; __syncthreads();
  for (int o = 128; o > 0; o >>= 1) { if (k < o) red[k] = fmaxf(red[k], red[k + o]); __syncthreads(); }
  const float mx = red[0]; __syncthreads();
  const float e = expf(v - mx);
  red[k] = e; __syncthreads();
  for (int o = 128; o > 0; o >>= 1) { if (k < o) red[k] += red[k + o]; __syncthreads(); }
  probs[rowo + k] = (_Float16)(e / red[0]);
}

// ------------------------- softmax: triangle attention ----------------------
// sc chunk [64][L][L] (batch = (i-i0)*4+h); tb [L*L,16]; probs f16 same shape.
__global__ void softmax_tri_kernel(const float* __restrict__ sc,
                                   const float* __restrict__ tb,
                                   const float* __restrict__ mask,
                                   _Float16* __restrict__ probs, int i0) {
  const int j = blockIdx.x, lb = blockIdx.y, k = threadIdx.x;
  const int i = i0 + (lb >> 2), h = lb & 3;
  const size_t rowo = ((size_t)lb * L_SEQ + j) * L_SEQ;
  float v = sc[rowo + k] + 1e9f * (mask[i] * mask[k] - 1.0f) +
            tb[((size_t)j * L_SEQ + k) * 16 + h];
  __shared__ float red[256];
  red[k] = v; __syncthreads();
  for (int o = 128; o > 0; o >>= 1) { if (k < o) red[k] = fmaxf(red[k], red[k + o]); __syncthreads(); }
  const float mx = red[0]; __syncthreads();
  const float e = expf(v - mx);
  red[k] = e; __syncthreads();
  for (int o = 128; o > 0; o >>= 1) { if (k < o) red[k] += red[k + o]; __syncthreads(); }
  probs[rowo + k] = (_Float16)(e / red[0]);
}

// ---------------------- elementwise / reshape kernels -----------------------

// qkv [L,3072] -> qb/kb [H=32][L][32] f16 (q scaled), vt [H][32][L] f16
__global__ void seq_gather_kernel(const float* __restrict__ qkv, _Float16* __restrict__ qb,
                                  _Float16* __restrict__ kb, _Float16* __restrict__ vt) {
  size_t i = (size_t)blockIdx.x * blockDim.x + threadIdx.x;
  if (i >= (size_t)32 * L_SEQ * 32) return;
  int c = (int)(i & 31);
  int l = (int)((i >> 5) % L_SEQ);
  int h = (int)(i / ((size_t)L_SEQ * 32));
  const size_t base = (size_t)l * 3072 + (size_t)h * 96;
  qb[((size_t)h * L_SEQ + l) * 32 + c] = (_Float16)(qkv[base + c] * QK_SCALE);
  kb[((size_t)h * L_SEQ + l) * 32 + c] = (_Float16)(qkv[base + 32 + c]);
  vt[((size_t)h * 32 + c) * L_SEQ + l] = (_Float16)(qkv[base + 64 + c]);
}

// q,k,v f32 [L*L,128] -> qb/kb [(i*4+h)][L][32], vt [(i*4+h)][32][L]
__global__ void tri_gather_kernel(const float* __restrict__ q, const float* __restrict__ k,
                                  const float* __restrict__ v, _Float16* __restrict__ qb,
                                  _Float16* __restrict__ kb, _Float16* __restrict__ vt) {
  size_t idx = (size_t)blockIdx.x * blockDim.x + threadIdx.x;
  if (idx >= (size_t)LLPAIR * CZ_DIM) return;
  int hc = (int)(idx & 127);
  size_t r = idx >> 7;
  int h = hc >> 5, c = hc & 31;
  int i = (int)(r / L_SEQ), j = (int)(r % L_SEQ);
  size_t bo = ((size_t)i * 4 + h);
  qb[(bo * L_SEQ + j) * 32 + c] = (_Float16)(q[idx] * QK_SCALE);
  kb[(bo * L_SEQ + j) * 32 + c] = (_Float16)(k[idx]);
  vt[(bo * 32 + c) * L_SEQ + j] = (_Float16)(v[idx]);
}

// y2g16[l, h*32+c] = gs[l, h*32+c] * y2b[h][l][c]   (gs already sigmoided)
__global__ void seq_combine_kernel(const float* __restrict__ gs, const float* __restrict__ y2b,
                                   _Float16* __restrict__ out) {
  size_t i = (size_t)blockIdx.x * blockDim.x + threadIdx.x;
  if (i >= (size_t)L_SEQ * CS_DIM) return;
  int hc = (int)(i % CS_DIM);
  int l = (int)(i / CS_DIM);
  int h = hc >> 5, c = hc & 31;
  out[i] = (_Float16)(gs[i] * y2b[((size_t)h * L_SEQ + l) * 32 + c]);
}

// o16[(i,j), h*32+c] = gs[(i,j),h*32+c] * ob[(i*4+h)][j][c]
__global__ void tri_combine_kernel(const float* __restrict__ gs, const float* __restrict__ ob,
                                   _Float16* __restrict__ out) {
  size_t idx = (size_t)blockIdx.x * blockDim.x + threadIdx.x;
  if (idx >= (size_t)LLPAIR * CZ_DIM) return;
  int hc = (int)(idx & 127);
  size_t r = idx >> 7;
  int h = hc >> 5, c = hc & 31;
  int i = (int)(r / L_SEQ), j = (int)(r % L_SEQ);
  out[idx] = (_Float16)(gs[idx] * ob[(((size_t)i * 4 + h) * L_SEQ + j) * 32 + c]);
}

// a16 = mask[i]*mask[j] * gs * pa      (gs already sigmoided)
__global__ void sigmul_kernel(const float* __restrict__ gs, const float* __restrict__ pa,
                              const float* __restrict__ mask, _Float16* __restrict__ out) {
  size_t idx = (size_t)blockIdx.x * blockDim.x + threadIdx.x;
  if (idx >= (size_t)LLPAIR * CZ_DIM) return;
  size_t r = idx >> 7;
  int i = (int)(r / L_SEQ), j = (int)(r % L_SEQ);
  out[idx] = (_Float16)(mask[i] * mask[j] * gs[idx] * pa[idx]);
}

// Z += tz * gz
__global__ void gatez_add_kernel(float* __restrict__ Z, const float* __restrict__ tz,
                                 const float* __restrict__ gz) {
  size_t idx = (size_t)blockIdx.x * blockDim.x + threadIdx.x;
  if (idx >= (size_t)LLPAIR * CZ_DIM) return;
  Z[idx] += tz[idx] * gz[idx];
}

// Z[(i,j)] += v[(i,j)] or v[(j,i)]
__global__ void residual_add_swap_kernel(float* __restrict__ Z, const float* __restrict__ v,
                                         int swap) {
  size_t idx = (size_t)blockIdx.x * blockDim.x + threadIdx.x;
  if (idx >= (size_t)LLPAIR * CZ_DIM) return;
  int c = (int)(idx & 127);
  size_t r = idx >> 7;
  int i = (int)(r / L_SEQ), j = (int)(r % L_SEQ);
  size_t src = swap ? (((size_t)j * L_SEQ + i) * CZ_DIM + c) : idx;
  Z[idx] += v[src];
}

// triangle transpose: at[c][r][col] = a16[(pair)*128 + c], pair = (r,col) or (col,r)
__global__ void tri_transpose_kernel(const _Float16* __restrict__ a16,
                                     _Float16* __restrict__ at, int swapIK) {
  size_t idx = (size_t)blockIdx.x * blockDim.x + threadIdx.x;
  if (idx >= (size_t)CZ_DIM * LLPAIR) return;
  int c = (int)(idx / LLPAIR);
  size_t rc = idx % LLPAIR;
  int r = (int)(rc / L_SEQ), col = (int)(rc % L_SEQ);
  size_t pr = swapIK ? ((size_t)col * L_SEQ + r) : ((size_t)r * L_SEQ + col);
  at[idx] = a16[pr * CZ_DIM + c];
}

// s2p: concat16[(i,j),c] = c<64 ? q_[j,c]*k_[i,c] : q_[j,c-64]-k_[i,c-64]
__global__ void build_concat_kernel(const float* __restrict__ qk, _Float16* __restrict__ out) {
  size_t idx = (size_t)blockIdx.x * blockDim.x + threadIdx.x;
  if (idx >= (size_t)LLPAIR * CZ_DIM) return;
  int c = (int)(idx & 127);
  size_t r = idx >> 7;
  int i = (int)(r / L_SEQ), j = (int)(r % L_SEQ);
  int cc = c & 63;
  float qv = qk[(size_t)j * 128 + cc];
  float kv = qk[(size_t)i * 128 + 64 + cc];
  out[idx] = (_Float16)((c < 64) ? (qv * kv) : (qv - kv));
}

// ------------------------------- host helpers ------------------------------

static inline void run_gemm(hipStream_t st, const _Float16* A, size_t sA,
                            const _Float16* Bt, size_t sB, const float* bias,
                            const float* addsrc, float* C, _Float16* C16, size_t sC,
                            int M, int N, int K, int batch, int act) {
  dim3 g((N + 63) / 64, (M + 63) / 64, batch), b(128);
  gemm_wmma_kernel<<<g, b, 0, st>>>(A, sA, Bt, sB, bias, addsrc, C, C16, sC, M, N, K, act);
}
static inline void run_wconv(hipStream_t st, const float* W, _Float16* Wt, int K, int N, int Npad) {
  size_t total = (size_t)Npad * K;
  wconv_kernel<<<dim3((unsigned)((total + 255) / 256)), dim3(256), 0, st>>>(W, Wt, K, N, Npad, total);
}
static inline void run_ln(hipStream_t st, const float* X, const float* g, const float* b,
                          _Float16* Y, int D, int rows, int swap, int chanMajor) {
  ln_kernel<<<dim3(rows), dim3(128), 0, st>>>(X, g, b, Y, D, rows, L_SEQ, swap, chanMajor);
}
static inline dim3 elgrid(size_t n) { return dim3((unsigned)((n + 255) / 256)); }

// ------------------------------- orchestration -----------------------------

extern "C" void kernel_launch(void* const* d_in, const int* in_sizes, int n_in,
                              void* d_out, int out_size, void* d_ws, size_t ws_size,
                              hipStream_t stream) {
  (void)in_sizes; (void)n_in; (void)out_size; (void)ws_size;
  const float* s_in = (const float*)d_in[0];
  const float* z_in = (const float*)d_in[1];
  const float* mask = (const float*)d_in[2];
  const float* P[80];
  for (int i = 0; i < 80; ++i) P[i] = (const float*)d_in[3 + i];
  // Param order (dict insertion order of _params):
  // 0 ln1_g 1 ln1_b 2 p2s_ln_g 3 p2s_ln_b 4 p2s_w 5 attn_wqkv 6 attn_wg 7 attn_bg
  // 8 attn_wo 9 attn_bo 10 s2p_ln_g 11 s2p_ln_b 12 s2p_wp 13 s2p_bp 14 s2p_wo 15 s2p_bo
  // 16.. tmo{lni_g,lni_b,lno_g,lno_b,wap,bap,wag,bag,wbp,bbp,wbg,bbg,wg,bg,wz,bz}
  // 32.. tmi{...}; 48.. tas{ln_g,ln_b,wb,wq,wk,wv,wg,bg,wo,bo}; 58.. tae{...}
  // 68.. mseq{ln_g,ln_b,w1,b1,w2,b2}; 74.. mpair{ln_g,ln_b,w1,b1,w2,b2}

  char* ws = (char*)d_ws;
  float*    Z     = (float*)(ws + 0);              // 33.55 MB
  float*    S     = (float*)(ws + 33554432);       // 1 MB
  _Float16* X16   = (_Float16*)(ws + 34603008);    // 16.78 MB
  _Float16* A16   = (_Float16*)(ws + 51380224);    // 16.78 MB
  _Float16* B16   = (_Float16*)(ws + 68157440);    // 16.78 MB
  _Float16* AT16  = (_Float16*)(ws + 84934656);    // 16.78 MB
  _Float16* BT16  = (_Float16*)(ws + 101711872);   // 16.78 MB (also scores f32 arena)
  _Float16* PB16  = (_Float16*)(ws + 118489088);   // 8.39 MB
  float*    TB32  = (float*)(ws + 126877696);      // 4.19 MB
  float*    BIAS32= (float*)(ws + 131072000);      // 8.39 MB
  _Float16* W16   = (_Float16*)(ws + 139460608);   // 16.78 MB (weight staging)
  float*    F32A  = (float*)(ws + 156237824);      // 33.55 MB
  float*    F32B  = (float*)(ws + 189792256);      // 33.55 MB
  float*    F32C  = (float*)(ws + 223346688);      // 33.55 MB

  const size_t NZ = (size_t)LLPAIR * CZ_DIM;

  hipMemcpyAsync(S, s_in, (size_t)L_SEQ * CS_DIM * 4, hipMemcpyDeviceToDevice, stream);
  hipMemcpyAsync(Z, z_in, NZ * 4, hipMemcpyDeviceToDevice, stream);

  // ---- Stage 1: pair_to_sequence bias = LN(z) @ p2s_w  -> BIAS32 [LL,32]
  run_ln(stream, Z, P[2], P[3], X16, CZ_DIM, LLPAIR, 0, 0);
  run_wconv(stream, P[4], W16, CZ_DIM, 32, 32);
  run_gemm(stream, X16, 0, W16, 0, nullptr, nullptr, BIAS32, nullptr, 0,
           LLPAIR, 32, CZ_DIM, 1, 0);

  // ---- Stage 2: gated sequence attention with pair bias
  run_ln(stream, S, P[0], P[1], X16, CS_DIM, L_SEQ, 0, 0);          // y16
  run_wconv(stream, P[5], W16, CS_DIM, 3 * CS_DIM, 3 * CS_DIM);
  run_gemm(stream, X16, 0, W16, 0, nullptr, nullptr, F32A, nullptr, 0,
           L_SEQ, 3 * CS_DIM, CS_DIM, 1, 0);                         // qkv
  seq_gather_kernel<<<elgrid((size_t)32 * L_SEQ * 32), 256, 0, stream>>>(F32A, A16, B16, AT16);
  run_gemm(stream, A16, (size_t)L_SEQ * 32, B16, (size_t)L_SEQ * 32, nullptr, nullptr,
           F32B, nullptr, (size_t)L_SEQ * L_SEQ, L_SEQ, L_SEQ, 32, 32, 0);   // scores[h]
  softmax_seq_kernel<<<dim3(L_SEQ, 32), 256, 0, stream>>>(F32B, BIAS32, mask, PB16);
  run_gemm(stream, PB16, (size_t)L_SEQ * L_SEQ, AT16, (size_t)32 * L_SEQ, nullptr, nullptr,
           F32C, nullptr, (size_t)L_SEQ * 32, L_SEQ, 32, L_SEQ, 32, 0);      // y2_b[h]
  run_wconv(stream, P[6], W16, CS_DIM, CS_DIM, CS_DIM);
  run_gemm(stream, X16, 0, W16, 0, P[7], nullptr, F32A, nullptr, 0,
           L_SEQ, CS_DIM, CS_DIM, 1, 2);                             // sigmoid gate
  seq_combine_kernel<<<elgrid((size_t)L_SEQ * CS_DIM), 256, 0, stream>>>(F32A, F32C, B16);
  run_wconv(stream, P[8], W16, CS_DIM, CS_DIM, CS_DIM);
  run_gemm(stream, B16, 0, W16, 0, P[9], S, S, nullptr, 0,
           L_SEQ, CS_DIM, CS_DIM, 1, 0);                             // s += ...

  // ---- Stage 3: mseq MLP
  run_ln(stream, S, P[68], P[69], X16, CS_DIM, L_SEQ, 0, 0);
  run_wconv(stream, P[70], W16, CS_DIM, 4 * CS_DIM, 4 * CS_DIM);
  run_gemm(stream, X16, 0, W16, 0, P[71], nullptr, nullptr, B16, 0,
           L_SEQ, 4 * CS_DIM, CS_DIM, 1, 1);                         // relu -> f16
  run_wconv(stream, P[72], W16, 4 * CS_DIM, CS_DIM, CS_DIM);
  run_gemm(stream, B16, 0, W16, 0, P[73], S, S, nullptr, 0,
           L_SEQ, CS_DIM, 4 * CS_DIM, 1, 0);

  // ---- Stage 4: sequence_to_pair
  run_ln(stream, S, P[10], P[11], X16, CS_DIM, L_SEQ, 0, 0);
  run_wconv(stream, P[12], W16, CS_DIM, 128, 128);
  run_gemm(stream, X16, 0, W16, 0, P[13], nullptr, F32A, nullptr, 0,
           L_SEQ, 128, CS_DIM, 1, 0);                                // qk [L,128]
  build_concat_kernel<<<elgrid(NZ), 256, 0, stream>>>(F32A, A16);
  run_wconv(stream, P[14], W16, 128, CZ_DIM, CZ_DIM);
  run_gemm(stream, A16, 0, W16, 0, P[15], Z, Z, nullptr, 0,
           LLPAIR, CZ_DIM, 128, 1, 0);                               // z += ...

  // ---- Stages 5/6: triangular multiplicative updates (outgoing, incoming)
  for (int m = 0; m < 2; ++m) {
    const int pb = 16 + 16 * m;
    const int swapIK = m;  // 0 = outgoing (bikc,bjkc), 1 = incoming (bkic,bkjc)
    run_ln(stream, Z, P[pb + 0], P[pb + 1], X16, CZ_DIM, LLPAIR, 0, 0);
    run_wconv(stream, P[pb + 4], W16, CZ_DIM, CZ_DIM, CZ_DIM);       // wap
    run_gemm(stream, X16, 0, W16, 0, P[pb + 5], nullptr, F32A, nullptr, 0,
             LLPAIR, CZ_DIM, CZ_DIM, 1, 0);
    run_wconv(stream, P[pb + 6], W16, CZ_DIM, CZ_DIM, CZ_DIM);       // wag
    run_gemm(stream, X16, 0, W16, 0, P[pb + 7], nullptr, F32B, nullptr, 0,
             LLPAIR, CZ_DIM, CZ_DIM, 1, 2);
    sigmul_kernel<<<elgrid(NZ), 256, 0, stream>>>(F32B, F32A, mask, A16);
    run_wconv(stream, P[pb + 8], W16, CZ_DIM, CZ_DIM, CZ_DIM);       // wbp
    run_gemm(stream, X16, 0, W16, 0, P[pb + 9], nullptr, F32A, nullptr, 0,
             LLPAIR, CZ_DIM, CZ_DIM, 1, 0);
    run_wconv(stream, P[pb + 10], W16, CZ_DIM, CZ_DIM, CZ_DIM);      // wbg
    run_gemm(stream, X16, 0, W16, 0, P[pb + 11], nullptr, F32B, nullptr, 0,
             LLPAIR, CZ_DIM, CZ_DIM, 1, 2);
    sigmul_kernel<<<elgrid(NZ), 256, 0, stream>>>(F32B, F32A, mask, B16);
    tri_transpose_kernel<<<elgrid(NZ), 256, 0, stream>>>(A16, AT16, swapIK);
    tri_transpose_kernel<<<elgrid(NZ), 256, 0, stream>>>(B16, BT16, swapIK);
    // 128 per-channel GEMMs: t_t[c] = A_c (256x256) * B_c^T
    run_gemm(stream, AT16, (size_t)LLPAIR, BT16, (size_t)LLPAIR, nullptr, nullptr,
             F32A, nullptr, (size_t)LLPAIR, L_SEQ, L_SEQ, L_SEQ, CZ_DIM, 0);
    run_ln(stream, F32A, P[pb + 2], P[pb + 3], B16, CZ_DIM, LLPAIR, 0, 1);  // chan-major in
    run_wconv(stream, P[pb + 14], W16, CZ_DIM, CZ_DIM, CZ_DIM);      // wz
    run_gemm(stream, B16, 0, W16, 0, P[pb + 15], nullptr, F32B, nullptr, 0,
             LLPAIR, CZ_DIM, CZ_DIM, 1, 0);
    run_wconv(stream, P[pb + 12], W16, CZ_DIM, CZ_DIM, CZ_DIM);      // wg
    run_gemm(stream, X16, 0, W16, 0, P[pb + 13], nullptr, F32C, nullptr, 0,
             LLPAIR, CZ_DIM, CZ_DIM, 1, 2);
    gatez_add_kernel<<<elgrid(NZ), 256, 0, stream>>>(Z, F32B, F32C);
  }

  // ---- Stages 7/8: triangular attention (starting, ending)
  for (int m = 0; m < 2; ++m) {
    const int pb = 48 + 10 * m;
    const int sw = m;  // ending node: i<->j transpose at entry/exit
    run_ln(stream, Z, P[pb + 0], P[pb + 1], X16, CZ_DIM, LLPAIR, sw, 0);
    run_wconv(stream, P[pb + 2], W16, CZ_DIM, 4, 16);                // wb (pad N 4->16)
    run_gemm(stream, X16, 0, W16, 0, nullptr, nullptr, TB32, nullptr, 0,
             LLPAIR, 16, CZ_DIM, 1, 0);                              // tb [LL,16]
    run_wconv(stream, P[pb + 3], W16, CZ_DIM, CZ_DIM, CZ_DIM);
    run_gemm(stream, X16, 0, W16, 0, nullptr, nullptr, F32A, nullptr, 0,
             LLPAIR, CZ_DIM, CZ_DIM, 1, 0);                          // q
    run_wconv(stream, P[pb + 4], W16, CZ_DIM, CZ_DIM, CZ_DIM);
    run_gemm(stream, X16, 0, W16, 0, nullptr, nullptr, F32B, nullptr, 0,
             LLPAIR, CZ_DIM, CZ_DIM, 1, 0);                          // k
    run_wconv(stream, P[pb + 5], W16, CZ_DIM, CZ_DIM, CZ_DIM);
    run_gemm(stream, X16, 0, W16, 0, nullptr, nullptr, F32C, nullptr, 0,
             LLPAIR, CZ_DIM, CZ_DIM, 1, 0);                          // v
    tri_gather_kernel<<<elgrid(NZ), 256, 0, stream>>>(F32A, F32B, F32C, A16, B16, AT16);
    run_wconv(stream, P[pb + 6], W16, CZ_DIM, CZ_DIM, CZ_DIM);       // wg
    run_gemm(stream, X16, 0, W16, 0, P[pb + 7], nullptr, F32A, nullptr, 0,
             LLPAIR, CZ_DIM, CZ_DIM, 1, 2);                          // sigmoid gate
    // attention, chunked over i (16 i rows -> 64 batches per chunk)
    float* SCORES = (float*)BT16;  // 64 * 65536 f32 = 16.78 MB, reuses BT arena
    for (int ic = 0; ic < 16; ++ic) {
      const size_t boff = (size_t)ic * 64;
      run_gemm(stream, A16 + boff * ((size_t)L_SEQ * 32), (size_t)L_SEQ * 32,
               B16 + boff * ((size_t)L_SEQ * 32), (size_t)L_SEQ * 32, nullptr, nullptr,
               SCORES, nullptr, (size_t)LLPAIR, L_SEQ, L_SEQ, 32, 64, 0);
      softmax_tri_kernel<<<dim3(L_SEQ, 64), 256, 0, stream>>>(SCORES, TB32, mask, PB16, ic * 16);
      run_gemm(stream, PB16, (size_t)LLPAIR,
               AT16 + boff * ((size_t)32 * L_SEQ), (size_t)32 * L_SEQ, nullptr, nullptr,
               F32B + boff * ((size_t)L_SEQ * 32), nullptr, (size_t)L_SEQ * 32,
               L_SEQ, 32, L_SEQ, 64, 0);                             // o_b chunk
    }
    tri_combine_kernel<<<elgrid(NZ), 256, 0, stream>>>(F32A, F32B, X16);  // gated o -> f16
    run_wconv(stream, P[pb + 8], W16, CZ_DIM, CZ_DIM, CZ_DIM);       // wo
    run_gemm(stream, X16, 0, W16, 0, P[pb + 9], nullptr, F32C, nullptr, 0,
             LLPAIR, CZ_DIM, CZ_DIM, 1, 0);
    residual_add_swap_kernel<<<elgrid(NZ), 256, 0, stream>>>(Z, F32C, sw);
  }

  // ---- Stage 9: mpair MLP (row-chunked to bound scratch)
  run_ln(stream, Z, P[74], P[75], X16, CZ_DIM, LLPAIR, 0, 0);
  run_wconv(stream, P[76], W16, CZ_DIM, 512, 512);                   // w1t
  run_wconv(stream, P[78], W16 + (size_t)512 * CZ_DIM, 512, CZ_DIM, CZ_DIM);  // w2t
  for (int rc = 0; rc < 4; ++rc) {
    const size_t roff = (size_t)rc * 16384;
    run_gemm(stream, X16 + roff * CZ_DIM, 0, W16, 0, P[77], nullptr, nullptr, BT16, 0,
             16384, 512, CZ_DIM, 1, 1);                              // relu -> f16
    run_gemm(stream, BT16, 0, W16 + (size_t)512 * CZ_DIM, 0, P[79],
             Z + roff * CZ_DIM, Z + roff * CZ_DIM, nullptr, 0,
             16384, CZ_DIM, 512, 1, 0);
  }

  // ---- outputs: (s, z) concatenated flat
  hipMemcpyAsync((float*)d_out, S, (size_t)L_SEQ * CS_DIM * 4, hipMemcpyDeviceToDevice, stream);
  hipMemcpyAsync((float*)d_out + (size_t)L_SEQ * CS_DIM, Z, NZ * 4,
                 hipMemcpyDeviceToDevice, stream);
}
